// GAT_1005022347907
// MI455X (gfx1250) — matrix operand order repaired and checked by reading, not compiled, not run on hardware
//
#include <hip/hip_runtime.h>
#include <cstddef>

typedef __attribute__((ext_vector_type(16))) __bf16 v16bf;
typedef __attribute__((ext_vector_type(8)))  float  v8f;

#define NEG_SLOPE 0.2f

// ---------- helpers ----------
__device__ __forceinline__ unsigned pack_bf16(float a, float b) {
    unsigned ua = __float_as_uint(a), ub = __float_as_uint(b);
    unsigned ha = (ua + 0x7FFFu + ((ua >> 16) & 1u)) >> 16;   // RNE f32->bf16
    unsigned hb = (ub + 0x7FFFu + ((ub >> 16) & 1u)) >> 16;
    return (ha & 0xFFFFu) | (hb << 16);
}
// monotone float<->uint encoding so atomicMax(uint) == float max
__device__ __forceinline__ unsigned f2ord(float f) {
    unsigned u = __float_as_uint(f);
    return (u & 0x80000000u) ? ~u : (u | 0x80000000u);
}
__device__ __forceinline__ float ord2f(unsigned u) {
    return (u & 0x80000000u) ? __uint_as_float(u & 0x7FFFFFFFu)
                             : __uint_as_float(~u);
}
#define ORD_NEG_INF 0x007FFFFFu   // f2ord(-inf)

// ---------- bf16 WMMA GEMM: C[M,Ncol] = A[M,K(lda)] * B[K,Ncol] ----------
// Block: 256 threads (8 waves, wave32). Block tile: 256 rows x Ncol.
// Each wave: 32 rows (2 x 16-row fragments) x NT 16-col tiles.
// LDS rows padded to 20 dwords: keeps 16B alignment for ds_load_b128 and
// spreads rows across banks.
template<int NT>
__global__ __launch_bounds__(256)
void gemm_bf16_wmma(const float* __restrict__ A, const float* __restrict__ B,
                    float* __restrict__ C, int M, int K, int Ncol, int lda)
{
    __shared__ __align__(16) unsigned Ap[256][20];   // Ap[m][k/2] bf16 pairs
    __shared__ __align__(16) unsigned Bpt[128][20];  // Bpt[n][k/2] (B transposed)

    const int t    = threadIdx.x;
    const int lane = t & 31;
    const int wave = t >> 5;
    const int row0 = blockIdx.x * 256;
    const int m0   = wave * 32;
    const int hsel = lane >> 4;        // lane half: 0 or 1
    const int lid  = lane & 15;

    union Frag { unsigned u[8]; uint4 q[2]; v16bf v; };

    v8f acc[2][NT];
    #pragma unroll
    for (int rt = 0; rt < 2; ++rt)
        #pragma unroll
        for (int i = 0; i < NT; ++i)
            #pragma unroll
            for (int j = 0; j < 8; ++j) acc[rt][i][j] = 0.0f;

    for (int k0 = 0; k0 < K; k0 += 32) {
        // ---- stage A tile: 256 rows x 32 K (f32 -> packed bf16), 1 row/thread ----
        {
            const int ro = row0 + t;
            float vals[32];
            if (ro < M && k0 + 32 <= K) {
                const float4* p = (const float4*)(A + (size_t)ro * lda + k0);
                #pragma unroll
                for (int q = 0; q < 8; ++q) {
                    float4 f = p[q];
                    vals[4*q+0] = f.x; vals[4*q+1] = f.y;
                    vals[4*q+2] = f.z; vals[4*q+3] = f.w;
                }
            } else {
                #pragma unroll
                for (int q = 0; q < 32; ++q) {
                    int kk = k0 + q;
                    vals[q] = (ro < M && kk < K) ? A[(size_t)ro * lda + kk] : 0.0f;
                }
            }
            unsigned pk[16];
            #pragma unroll
            for (int j = 0; j < 16; ++j) pk[j] = pack_bf16(vals[2*j], vals[2*j+1]);
            uint4* dst = (uint4*)&Ap[t][0];
            #pragma unroll
            for (int j = 0; j < 4; ++j)
                dst[j] = uint4{pk[4*j], pk[4*j+1], pk[4*j+2], pk[4*j+3]};
        }
        // ---- stage B tile transposed: Bpt[n][kp], zero-padded to 128 cols ----
        {
            const int n = t & 127;
            const int g = t >> 7;      // 0 or 1 -> kpairs [0..7] / [8..15]
            unsigned pk[8];
            #pragma unroll
            for (int j = 0; j < 8; ++j) {
                int kp  = g * 8 + j;
                int kr0 = k0 + 2 * kp, kr1 = kr0 + 1;
                float b0 = (n < Ncol && kr0 < K) ? B[(size_t)kr0 * Ncol + n] : 0.0f;
                float b1 = (n < Ncol && kr1 < K) ? B[(size_t)kr1 * Ncol + n] : 0.0f;
                pk[j] = pack_bf16(b0, b1);
            }
            uint4* dst = (uint4*)&Bpt[n][g * 8];
            dst[0] = uint4{pk[0], pk[1], pk[2], pk[3]};
            dst[1] = uint4{pk[4], pk[5], pk[6], pk[7]};
        }
        __syncthreads();

        // ---- A fragments: 2 x (2 x ds_load_b128) ----
        Frag a0, a1;
        a0.q[0] = *(const uint4*)&Ap[m0 + lid][hsel * 4];
        a0.q[1] = *(const uint4*)&Ap[m0 + lid][8 + hsel * 4];
        a1.q[0] = *(const uint4*)&Ap[m0 + 16 + lid][hsel * 4];
        a1.q[1] = *(const uint4*)&Ap[m0 + 16 + lid][8 + hsel * 4];

        // ---- column strip: each B fragment feeds two WMMAs ----
        #pragma unroll
        for (int nt = 0; nt < NT; ++nt) {
            Frag bfr;
            bfr.q[0] = *(const uint4*)&Bpt[nt * 16 + lid][hsel * 4];
            bfr.q[1] = *(const uint4*)&Bpt[nt * 16 + lid][8 + hsel * 4];
            acc[0][nt] = __builtin_amdgcn_wmma_f32_16x16x32_bf16(
                false, a0.v, false, bfr.v, (short)0, acc[0][nt], false, false);
            acc[1][nt] = __builtin_amdgcn_wmma_f32_16x16x32_bf16(
                false, a1.v, false, bfr.v, (short)0, acc[1][nt], false, false);
        }
        __syncthreads();
    }

    // ---- store (C 16x16 f32 layout: M = v + 8*hsel, N = lid) ----
    #pragma unroll
    for (int rt = 0; rt < 2; ++rt) {
        #pragma unroll
        for (int nt = 0; nt < NT; ++nt) {
            int n = nt * 16 + lid;
            if (n >= Ncol) continue;
            #pragma unroll
            for (int v = 0; v < 8; ++v) {
                int m = row0 + m0 + rt * 16 + v + hsel * 8;
                if (m < M) C[(size_t)m * Ncol + n] = acc[rt][nt][v];
            }
        }
    }
}

// ---------- fills ----------
__global__ void k_fill_u32(unsigned* p, unsigned v, int n) {
    int i = blockIdx.x * blockDim.x + threadIdx.x;
    if (i < n) p[i] = v;
}

// ---------- attention projections: a_src/a_dst = <h[n,h,:], att[h,:]> ----------
__global__ void k_att(const float* __restrict__ h, const float* __restrict__ att_s,
                      const float* __restrict__ att_d, float* as_, float* ad_,
                      int N, int H, int C)
{
    int i = blockIdx.x * blockDim.x + threadIdx.x;
    if (i >= N * H) return;
    int n = i / H, hh = i % H;
    const float* row = h + (size_t)n * H * C + (size_t)hh * C;
    float s = 0.f, d = 0.f;
    for (int c = 0; c < C; ++c) {
        float v = row[c];
        s += v * att_s[hh * C + c];
        d += v * att_d[hh * C + c];
    }
    as_[i] = s; ad_[i] = d;
}

// ---------- segment max over incoming edges (ordered-uint atomicMax) ----------
__global__ void k_edge_max(const int* __restrict__ eidx,
                           const float* __restrict__ as_, const float* __restrict__ ad_,
                           unsigned* __restrict__ m_enc, int E, int N, int H)
{
    int e = blockIdx.x * blockDim.x + threadIdx.x;
    if (e >= E + N) return;
    int s, d;
    if (e < E) { s = eidx[e]; d = eidx[E + e]; } else { s = d = e - E; }
    if ((unsigned)s >= (unsigned)N || (unsigned)d >= (unsigned)N) return;
    for (int hh = 0; hh < H; ++hh) {
        float v = as_[s * H + hh] + ad_[d * H + hh];
        v = (v > 0.f) ? v : NEG_SLOPE * v;
        atomicMax(&m_enc[d * H + hh], f2ord(v));
    }
}

// ---------- exp(e - max) per edge + segment-sum of denominators ----------
__global__ void k_edge_exp(const int* __restrict__ eidx,
                           const float* __restrict__ as_, const float* __restrict__ ad_,
                           const unsigned* __restrict__ m_enc,
                           float* __restrict__ ex, float* __restrict__ denom,
                           int E, int N, int H)
{
    int e = blockIdx.x * blockDim.x + threadIdx.x;
    if (e >= E + N) return;
    int s, d;
    if (e < E) { s = eidx[e]; d = eidx[E + e]; } else { s = d = e - E; }
    if ((unsigned)s >= (unsigned)N || (unsigned)d >= (unsigned)N) return;
    for (int hh = 0; hh < H; ++hh) {
        float v = as_[s * H + hh] + ad_[d * H + hh];
        v = (v > 0.f) ? v : NEG_SLOPE * v;
        float xv = __expf(v - ord2f(m_enc[d * H + hh]));
        ex[(size_t)e * H + hh] = xv;
        atomicAdd(&denom[d * H + hh], xv);
    }
}

// ---------- weighted scatter-add: one thread per (edge, channel) ----------
__global__ void k_aggr(const float* __restrict__ hfeat, const float* __restrict__ ex,
                       const float* __restrict__ denom, const int* __restrict__ eidx,
                       float* __restrict__ out, int E, int N, int H, int C)
{
    int HC = H * C;
    int w = blockIdx.x * blockDim.x + threadIdx.x;
    long long total = (long long)(E + N) * HC;
    if ((long long)w >= total) return;
    int e = w / HC, r = w % HC, head = r / C;
    int s, d;
    if (e < E) { s = eidx[e]; d = eidx[E + e]; } else { s = d = e - E; }
    if ((unsigned)s >= (unsigned)N || (unsigned)d >= (unsigned)N) return;
    float alpha = ex[(size_t)e * H + head] / denom[(size_t)d * H + head];
    atomicAdd(&out[(size_t)d * HC + r], hfeat[(size_t)s * HC + r] * alpha);
}

// ---------- bias (+ optional ReLU) ----------
__global__ void k_bias(float* __restrict__ out, const float* __restrict__ bias,
                       int total, int HC, int relu)
{
    int i = blockIdx.x * blockDim.x + threadIdx.x;
    if (i >= total) return;
    float v = out[i] + bias[i % HC];
    if (relu) v = fmaxf(v, 0.0f);
    out[i] = v;
}

// ---------- launcher ----------
extern "C" void kernel_launch(void* const* d_in, const int* in_sizes, int n_in,
                              void* d_out, int out_size, void* d_ws, size_t ws_size,
                              hipStream_t stream)
{
    const int F_in = 500, H1 = 4, C1 = 32, HC1 = 128, F_out = 40;
    const float* x        = (const float*)d_in[0];
    const int*   eidx     = (const int*)d_in[1];
    const float* W1       = (const float*)d_in[2];
    const float* att_src1 = (const float*)d_in[3];
    const float* att_dst1 = (const float*)d_in[4];
    const float* b1       = (const float*)d_in[5];
    const float* W2       = (const float*)d_in[6];
    const float* att_src2 = (const float*)d_in[7];
    const float* att_dst2 = (const float*)d_in[8];
    const float* b2       = (const float*)d_in[9];
    float* out = (float*)d_out;

    const int N  = in_sizes[0] / F_in;
    const int E  = in_sizes[1] / 2;
    const int EE = E + N;

    // workspace carve-out (floats / u32, all 4-byte)
    float* ws = (float*)d_ws;
    size_t o = 0;
    float*    h1     = ws + o; o += (size_t)N * HC1;
    float*    out1   = ws + o; o += (size_t)N * HC1;
    float*    as1    = ws + o; o += (size_t)N * H1;
    float*    ad1    = ws + o; o += (size_t)N * H1;
    unsigned* m1     = (unsigned*)(ws + o); o += (size_t)N * H1;
    float*    den1   = ws + o; o += (size_t)N * H1;
    float*    ex1    = ws + o; o += (size_t)EE * H1;
    float*    h2     = ws + o; o += (size_t)N * F_out;
    float*    as2    = ws + o; o += (size_t)N;
    float*    ad2    = ws + o; o += (size_t)N;
    unsigned* m2     = (unsigned*)(ws + o); o += (size_t)N;
    float*    den2   = ws + o; o += (size_t)N;
    float*    ex2    = ws + o; o += (size_t)EE;
    (void)ws_size; (void)n_in; (void)out_size;

    const int TB = 256;
    dim3 blk(TB);
    auto gridFor = [](long long n, int tb) { return dim3((unsigned)((n + tb - 1) / tb)); };

    // ===== Layer 1 =====
    k_fill_u32<<<gridFor((long long)N * H1, TB), blk, 0, stream>>>(m1, ORD_NEG_INF, N * H1);
    k_fill_u32<<<gridFor((long long)N * H1, TB), blk, 0, stream>>>((unsigned*)den1, 0u, N * H1);
    k_fill_u32<<<gridFor((long long)N * HC1, TB), blk, 0, stream>>>((unsigned*)out1, 0u, N * HC1);

    gemm_bf16_wmma<8><<<dim3((N + 255) / 256), blk, 0, stream>>>(x, W1, h1, N, F_in, HC1, F_in);

    k_att<<<gridFor((long long)N * H1, TB), blk, 0, stream>>>(h1, att_src1, att_dst1, as1, ad1, N, H1, C1);
    k_edge_max<<<gridFor(EE, TB), blk, 0, stream>>>(eidx, as1, ad1, m1, E, N, H1);
    k_edge_exp<<<gridFor(EE, TB), blk, 0, stream>>>(eidx, as1, ad1, m1, ex1, den1, E, N, H1);
    k_aggr<<<gridFor((long long)EE * HC1, TB), blk, 0, stream>>>(h1, ex1, den1, eidx, out1, E, N, H1, C1);
    k_bias<<<gridFor((long long)N * HC1, TB), blk, 0, stream>>>(out1, b1, N * HC1, HC1, 1);

    // ===== Layer 2 =====
    k_fill_u32<<<gridFor(N, TB), blk, 0, stream>>>(m2, ORD_NEG_INF, N);
    k_fill_u32<<<gridFor(N, TB), blk, 0, stream>>>((unsigned*)den2, 0u, N);
    k_fill_u32<<<gridFor((long long)N * F_out, TB), blk, 0, stream>>>((unsigned*)out, 0u, N * F_out);

    gemm_bf16_wmma<3><<<dim3((N + 255) / 256), blk, 0, stream>>>(out1, W2, h2, N, HC1, F_out, HC1);

    k_att<<<gridFor(N, TB), blk, 0, stream>>>(h2, att_src2, att_dst2, as2, ad2, N, 1, F_out);
    k_edge_max<<<gridFor(EE, TB), blk, 0, stream>>>(eidx, as2, ad2, m2, E, N, 1);
    k_edge_exp<<<gridFor(EE, TB), blk, 0, stream>>>(eidx, as2, ad2, m2, ex2, den2, E, N, 1);
    k_aggr<<<gridFor((long long)EE * F_out, TB), blk, 0, stream>>>(h2, ex2, den2, eidx, out, E, N, 1, F_out);
    k_bias<<<gridFor((long long)N * F_out, TB), blk, 0, stream>>>(out, b2, N * F_out, F_out, 0);
}
